// SpaDCSA_32126355374630
// MI455X (gfx1250) — matrix operand order repaired and checked by reading, not compiled, not run on hardware
//
#include <hip/hip_runtime.h>
#include <hip/hip_bf16.h>

// Problem constants (from reference setup_inputs)
#define B_    2
#define C_    256
#define GROUPS 8
#define CPG   32          // channels per group == head dim d
#define H_    48
#define W_    48
#define HW    2304        // 48*48
#define EPS   1e-5f

typedef __attribute__((ext_vector_type(16))) __bf16 v16bf;
typedef __attribute__((ext_vector_type(8)))  __bf16 v8bf;
typedef __attribute__((ext_vector_type(8)))  float  v8f;
typedef __attribute__((ext_vector_type(4)))  unsigned int v4u;
typedef __attribute__((ext_vector_type(8)))  unsigned int v8u;

// ---------------------------------------------------------------------------
// Kernel 1: grouped QKV convolutions (Q: 3x3 pad1, K/V: 1x1), emit bf16.
// q,k stored [bg][n][d]; v stored transposed [bg][d][n].
// ---------------------------------------------------------------------------
__global__ __launch_bounds__(256) void qkv_kernel(
    const float* __restrict__ x,
    const float* __restrict__ Wq, const float* __restrict__ bq,
    const float* __restrict__ Wk, const float* __restrict__ bk,
    const float* __restrict__ Wv, const float* __restrict__ bv,
    __bf16* __restrict__ qws, __bf16* __restrict__ kws, __bf16* __restrict__ vtws)
{
    int idx = blockIdx.x * 256 + threadIdx.x;        // flat over [b, co, n]
    if (idx >= B_ * C_ * HW) return;
    int n  = idx % HW;
    int co = (idx / HW) % C_;
    int b  = idx / (HW * C_);
    int g  = co / CPG;
    int d  = co % CPG;
    int y  = n / W_, xx = n % W_;

    const float* xb = x + ((size_t)b * C_ + (size_t)g * CPG) * HW;  // group base

    float ak = bk[co], av = bv[co];
    const float* wkr = Wk + (size_t)co * CPG;
    const float* wvr = Wv + (size_t)co * CPG;
    #pragma unroll 4
    for (int ci = 0; ci < CPG; ++ci) {
        float xv_ = xb[(size_t)ci * HW + n];
        ak += xv_ * wkr[ci];
        av += xv_ * wvr[ci];
    }

    float aq = bq[co];
    const float* wqr = Wq + (size_t)co * CPG * 9;
    for (int ci = 0; ci < CPG; ++ci) {
        const float* xc = xb + (size_t)ci * HW;
        #pragma unroll
        for (int dy = -1; dy <= 1; ++dy) {
            int yy = y + dy;
            if (yy < 0 || yy >= H_) continue;
            #pragma unroll
            for (int dx = -1; dx <= 1; ++dx) {
                int xw = xx + dx;
                if (xw < 0 || xw >= W_) continue;
                aq += xc[yy * W_ + xw] * wqr[ci * 9 + (dy + 1) * 3 + (dx + 1)];
            }
        }
    }

    size_t bg = (size_t)b * GROUPS + g;
    qws[(bg * HW + n) * CPG + d]  = (__bf16)aq;
    kws[(bg * HW + n) * CPG + d]  = (__bf16)ak;
    vtws[(bg * CPG + d) * HW + n] = (__bf16)av;
}

// ---------------------------------------------------------------------------
// TDM helper: issue a tensor_load_to_lds for a (up to) 2D bf16 tile.
// Builds D# group0 (128b) + group1 (256b) per CDNA5 ISA.
// tile_dim1 == 0 -> 1D tile. Tracked on TENSORcnt of the issuing wave.
// ---------------------------------------------------------------------------
__device__ __forceinline__ void tdm_load_2d(unsigned int lds_off, const void* gptr,
                                            unsigned int dim0, unsigned int dim1,
                                            unsigned int tile0, unsigned int tile1,
                                            unsigned long long stride0)
{
    unsigned long long ga = (unsigned long long)(uintptr_t)gptr;
    v4u g0;
    g0[0] = 1u;                                   // count=1, user mode
    g0[1] = lds_off;                              // lds_addr (bytes)
    g0[2] = (unsigned int)ga;                     // global_addr[31:0]
    g0[3] = ((unsigned int)(ga >> 32) & 0x01FFFFFFu) | 0x80000000u; // addr[56:32] | type=2
    v8u g1;
    g1[0] = 0x00010000u;                          // workgroup_mask=0, data_size=1 (2B)
    g1[1] = (dim0 & 0xFFFFu) << 16;               // tensor_dim0[15:0]
    g1[2] = (dim0 >> 16) | ((dim1 & 0xFFFFu) << 16);  // dim0[31:16] | dim1[15:0]
    g1[3] = (dim1 >> 16) | (tile0 << 16);         // dim1[31:16] | tile_dim0
    g1[4] = tile1 & 0xFFFFu;                      // tile_dim1 | tile_dim2(=0)
    g1[5] = (unsigned int)stride0;                // tensor_dim0_stride[31:0]
    g1[6] = (unsigned int)((stride0 >> 32) & 0xFFFFu);  // stride[47:32] | dim1_stride lo(=0)
    g1[7] = 0u;                                   // dim1_stride hi
    asm volatile("tensor_load_to_lds %0, %1" :: "s"(g0), "s"(g1) : "memory");
}

// XOR-tree max over each 16-lane group using v_permlane16_b32 (VALU, no LDS).
__device__ __forceinline__ float redmax16(float v)
{
    unsigned int u, p;
    u = __builtin_bit_cast(unsigned int, v);
    p = __builtin_amdgcn_permlane16(u, u, 0x67452301u, 0xEFCDAB89u, false, false); // ^1
    v = fmaxf(v, __builtin_bit_cast(float, p));
    u = __builtin_bit_cast(unsigned int, v);
    p = __builtin_amdgcn_permlane16(u, u, 0x45670123u, 0xCDEF89ABu, false, false); // ^2
    v = fmaxf(v, __builtin_bit_cast(float, p));
    u = __builtin_bit_cast(unsigned int, v);
    p = __builtin_amdgcn_permlane16(u, u, 0x32107654u, 0xBA98FEDCu, false, false); // ^4
    v = fmaxf(v, __builtin_bit_cast(float, p));
    u = __builtin_bit_cast(unsigned int, v);
    p = __builtin_amdgcn_permlane16(u, u, 0xFEDCBA98u, 0x76543210u, false, false); // ^8
    v = fmaxf(v, __builtin_bit_cast(float, p));
    return v;
}

// ---------------------------------------------------------------------------
// Kernel 2: flash attention per (b,g). 8 waves/block; each wave owns a 16-row
// Q tile. 64-key chunks are TDM-staged (double-buffered) into LDS by wave 0;
// all waves consume them via ds loads. Online softmax in base-2 domain; row
// sums accumulated with a ones-column WMMA (no shuffle reduction). P relayout
// C->A goes through per-wave LDS with an explicit s_wait_dscnt.
// ---------------------------------------------------------------------------
__global__ __launch_bounds__(256) void attn_kernel(
    const __bf16* __restrict__ qws, const __bf16* __restrict__ kws,
    const __bf16* __restrict__ vtws,
    float* __restrict__ outp, float* __restrict__ csum, float* __restrict__ csq)
{
    __shared__ __bf16 ktile[2][64][32];    // 2 x 4KB: 64 keys x d=32
    __shared__ __bf16 vtile[2][32][64];    // 2 x 4KB: d=32 x 64 keys
    __shared__ __bf16 pT[8][16][64];       // per-wave P tile (16KB)

    const int lane = threadIdx.x & 31;
    const int wave = threadIdx.x >> 5;
    const int bg   = blockIdx.y;           // b*8+g
    const int b    = bg >> 3;
    const int g    = bg & 7;
    const int n0   = (blockIdx.x * 8 + wave) * 16;
    const int half = lane >> 4;
    const int l16  = lane & 15;
    const int off  = half ? 8 : 0;         // A-layout K offset for this half
    const int boff = half ? 16 : 0;        // K offset for this half (contiguous k = 16*half + s, A and B)

    const __bf16* qbase = qws  + (size_t)bg * HW * CPG;
    const __bf16* kbase = kws  + (size_t)bg * HW * CPG;
    const __bf16* vbase = vtws + (size_t)bg * CPG * HW;

    // Q tile in WMMA A layout
    const __bf16* qrow = qbase + (size_t)(n0 + l16) * CPG;
    v8bf qlo = *(const v8bf*)(qrow + boff);
    v8bf qhi = *(const v8bf*)(qrow + boff + 8);
    v16bf qa = __builtin_shufflevector(qlo, qhi,
        0,1,2,3,4,5,6,7,8,9,10,11,12,13,14,15);

    // ones B-tile for the row-sum WMMA
    v16bf ones;
    #pragma unroll
    for (int i = 0; i < 16; ++i) ones[i] = (__bf16)1.0f;

    v8f o0 = {}, o1 = {}, osum = {};
    float mrun[8];
    #pragma unroll
    for (int j = 0; j < 8; ++j) mrun[j] = -3.0e38f;

    const float k2 = 0.25506807187095354f; // (1/sqrt(32)) * log2(e)

    // Prologue: wave 0 TDM-loads chunk 0 into buffer 0
    if (wave == 0) {
        tdm_load_2d((unsigned int)(uintptr_t)&ktile[0][0][0], kbase,
                    HW * CPG, 1, 64 * CPG, 0, HW * CPG);
        tdm_load_2d((unsigned int)(uintptr_t)&vtile[0][0][0], vbase,
                    HW, CPG, 64, CPG, HW);
    }

    for (int mb = 0, it = 0; mb < HW; mb += 64, ++it) {
        const int cur = it & 1;
        if (wave == 0) __builtin_amdgcn_s_wait_tensorcnt(0);
        __syncthreads();                   // publish buf[cur]; prior reads of buf[cur^1] done
        if (wave == 0 && mb + 64 < HW) {
            tdm_load_2d((unsigned int)(uintptr_t)&ktile[cur ^ 1][0][0],
                        kbase + (size_t)(mb + 64) * CPG,
                        HW * CPG, 1, 64 * CPG, 0, HW * CPG);
            tdm_load_2d((unsigned int)(uintptr_t)&vtile[cur ^ 1][0][0],
                        vbase + (size_t)(mb + 64),
                        HW, CPG, 64, CPG, HW);
        }

        // ---- S = Q K^T for 64 keys (4 tiles), K B-operands from LDS
        v16bf kb0 = *(const v16bf*)&ktile[cur][ 0 + l16][boff];
        v16bf kb1 = *(const v16bf*)&ktile[cur][16 + l16][boff];
        v16bf kb2 = *(const v16bf*)&ktile[cur][32 + l16][boff];
        v16bf kb3 = *(const v16bf*)&ktile[cur][48 + l16][boff];
        v8f zero = {};
        v8f s0 = __builtin_amdgcn_wmma_f32_16x16x32_bf16(false, qa, false, kb0, (short)0, zero, false, false);
        v8f s1 = __builtin_amdgcn_wmma_f32_16x16x32_bf16(false, qa, false, kb1, (short)0, zero, false, false);
        v8f s2 = __builtin_amdgcn_wmma_f32_16x16x32_bf16(false, qa, false, kb2, (short)0, zero, false, false);
        v8f s3 = __builtin_amdgcn_wmma_f32_16x16x32_bf16(false, qa, false, kb3, (short)0, zero, false, false);

        // ---- online softmax (base-2), P -> per-wave LDS in bf16
        #pragma unroll
        for (int j = 0; j < 8; ++j) {
            float a0 = s0[j] * k2, a1 = s1[j] * k2;
            float a2 = s2[j] * k2, a3 = s3[j] * k2;
            float t = fmaxf(fmaxf(a0, a1), fmaxf(a2, a3));
            t = redmax16(t);                       // row max across 16 lanes
            float mn   = fmaxf(mrun[j], t);
            float corr = exp2f(mrun[j] - mn);
            mrun[j] = mn;
            float p0 = exp2f(a0 - mn), p1 = exp2f(a1 - mn);
            float p2 = exp2f(a2 - mn), p3 = exp2f(a3 - mn);
            o0[j]   *= corr;
            o1[j]   *= corr;
            osum[j] *= corr;
            int prow = half ? (j + 8) : j;         // C-layout row for this half
            pT[wave][prow][ 0 + l16] = (__bf16)p0;
            pT[wave][prow][16 + l16] = (__bf16)p1;
            pT[wave][prow][32 + l16] = (__bf16)p2;
            pT[wave][prow][48 + l16] = (__bf16)p3;
        }

        // wave-local LDS RAW before A-layout reload
        asm volatile("s_wait_dscnt 0" ::: "memory");

        v8bf plo0 = *(const v8bf*)&pT[wave][l16][boff];
        v8bf phi0 = *(const v8bf*)&pT[wave][l16][boff + 8];
        v16bf pa0 = __builtin_shufflevector(plo0, phi0,
            0,1,2,3,4,5,6,7,8,9,10,11,12,13,14,15);
        v8bf plo1 = *(const v8bf*)&pT[wave][l16][32 + boff];
        v8bf phi1 = *(const v8bf*)&pT[wave][l16][32 + boff + 8];
        v16bf pa1 = __builtin_shufflevector(plo1, phi1,
            0,1,2,3,4,5,6,7,8,9,10,11,12,13,14,15);

        // ---- O += P V (V B-operands from LDS) and row sums += P * 1
        v16bf vb00 = *(const v16bf*)&vtile[cur][ 0 + l16][ 0 + boff];
        v16bf vb01 = *(const v16bf*)&vtile[cur][ 0 + l16][32 + boff];
        v16bf vb10 = *(const v16bf*)&vtile[cur][16 + l16][ 0 + boff];
        v16bf vb11 = *(const v16bf*)&vtile[cur][16 + l16][32 + boff];

        o0   = __builtin_amdgcn_wmma_f32_16x16x32_bf16(false, pa0, false, vb00, (short)0, o0,   false, false);
        o0   = __builtin_amdgcn_wmma_f32_16x16x32_bf16(false, pa1, false, vb01, (short)0, o0,   false, false);
        o1   = __builtin_amdgcn_wmma_f32_16x16x32_bf16(false, pa0, false, vb10, (short)0, o1,   false, false);
        o1   = __builtin_amdgcn_wmma_f32_16x16x32_bf16(false, pa1, false, vb11, (short)0, o1,   false, false);
        osum = __builtin_amdgcn_wmma_f32_16x16x32_bf16(false, pa0, false, ones, (short)0, osum, false, false);
        osum = __builtin_amdgcn_wmma_f32_16x16x32_bf16(false, pa1, false, ones, (short)0, osum, false, false);
    }

    // ---- normalize by row sums (osum holds the row sum in every N column)
    #pragma unroll
    for (int j = 0; j < 8; ++j) {
        float inv = 1.0f / osum[j];
        o0[j] *= inv;
        o1[j] *= inv;
    }

    // ---- store pre-BN output: out[b][g*32+d][n]
    float* outb = outp + ((size_t)b * C_ + (size_t)g * CPG) * HW;
    #pragma unroll
    for (int j = 0; j < 8; ++j) {
        int nrow = n0 + (half ? (j + 8) : j);
        outb[(size_t)l16 * HW + nrow]        = o0[j];
        outb[(size_t)(16 + l16) * HW + nrow] = o1[j];
    }

    // ---- per-channel batch stats (channel = g*32 + N column; lane halves share)
    float sa0 = 0.f, qa0 = 0.f, sa1 = 0.f, qa1 = 0.f;
    #pragma unroll
    for (int j = 0; j < 8; ++j) {
        sa0 += o0[j];  qa0 += o0[j] * o0[j];
        sa1 += o1[j];  qa1 += o1[j] * o1[j];
    }
    sa0 += __shfl_xor(sa0, 16);  qa0 += __shfl_xor(qa0, 16);
    sa1 += __shfl_xor(sa1, 16);  qa1 += __shfl_xor(qa1, 16);
    if (half == 0) {
        atomicAdd(&csum[g * CPG + l16],      sa0);
        atomicAdd(&csq [g * CPG + l16],      qa0);
        atomicAdd(&csum[g * CPG + 16 + l16], sa1);
        atomicAdd(&csq [g * CPG + 16 + l16], qa1);
    }
}

// ---------------------------------------------------------------------------
// Kernel 3: BatchNorm (training-mode biased stats) + affine + ReLU
// ---------------------------------------------------------------------------
__global__ __launch_bounds__(256) void bn_relu_kernel(
    const float* __restrict__ outp, const float* __restrict__ csum,
    const float* __restrict__ csq,  const float* __restrict__ gamma,
    const float* __restrict__ beta, float* __restrict__ out)
{
    int i = blockIdx.x * 256 + threadIdx.x;
    if (i >= B_ * C_ * HW) return;
    int c = (i / HW) % C_;
    const float invN = 1.0f / (float)(B_ * HW);
    float mean = csum[c] * invN;
    float var  = csq[c] * invN - mean * mean;
    float v = (outp[i] - mean) * rsqrtf(var + EPS);
    v = v * gamma[c] + beta[c];
    out[i] = fmaxf(v, 0.0f);
}

__global__ void zero_kernel(float* p, int n)
{
    int i = blockIdx.x * 256 + threadIdx.x;
    if (i < n) p[i] = 0.0f;
}

// ---------------------------------------------------------------------------
extern "C" void kernel_launch(void* const* d_in, const int* in_sizes, int n_in,
                              void* d_out, int out_size, void* d_ws, size_t ws_size,
                              hipStream_t stream)
{
    const float* x     = (const float*)d_in[0];
    const float* Wq    = (const float*)d_in[1];
    const float* bq    = (const float*)d_in[2];
    const float* Wk    = (const float*)d_in[3];
    const float* bk    = (const float*)d_in[4];
    const float* Wv    = (const float*)d_in[5];
    const float* bv    = (const float*)d_in[6];
    const float* gamma = (const float*)d_in[7];
    const float* beta  = (const float*)d_in[8];
    float* outd = (float*)d_out;

    // Workspace layout (bytes): q | k | vT (bf16) | out_pre (f32) | stats
    const size_t QELEM = (size_t)B_ * GROUPS * HW * CPG;   // 1,179,648
    char* ws = (char*)d_ws;
    __bf16* qws = (__bf16*)(ws);
    __bf16* kws = (__bf16*)(ws + QELEM * 2);
    __bf16* vt  = (__bf16*)(ws + QELEM * 4);
    float*  outp = (float*)(ws + QELEM * 6);
    float*  csum = outp + (size_t)B_ * C_ * HW;
    float*  csq  = csum + C_;

    const int NTOT = B_ * C_ * HW;

    zero_kernel<<<2, 256, 0, stream>>>(csum, 2 * C_);
    qkv_kernel<<<(NTOT + 255) / 256, 256, 0, stream>>>(
        x, Wq, bq, Wk, bk, Wv, bv, qws, kws, vt);
    attn_kernel<<<dim3(HW / 128, B_ * GROUPS), 256, 0, stream>>>(
        qws, kws, vt, outp, csum, csq);
    bn_relu_kernel<<<(NTOT + 255) / 256, 256, 0, stream>>>(
        outp, csum, csq, gamma, beta, outd);
}
